// MolecularGNN_82016695484626
// MI455X (gfx1250) — compile-verified, matrix operand order checked by reading, not run on hardware
//
#include <hip/hip_runtime.h>

typedef __attribute__((ext_vector_type(8)))  __bf16 v8bf;
typedef __attribute__((ext_vector_type(16))) __bf16 v16bf;
typedef __attribute__((ext_vector_type(8)))  float  v8f;

#define HH 128
#define BN_EPS 1e-5f

// ---------------- elementwise / setup kernels ----------------

__global__ void k_fill(float* p, float v, long n) {
    long i = (long)blockIdx.x * blockDim.x + threadIdx.x;
    if (i < n) p[i] = v;
}

__global__ void k_deg(const int* __restrict__ dst, float* __restrict__ deg, int E) {
    int e = blockIdx.x * blockDim.x + threadIdx.x;
    if (e < E) atomicAdd(&deg[dst[e]], 1.0f);
}

__global__ void k_rsqrt_inplace(float* p, int n) {
    int i = blockIdx.x * blockDim.x + threadIdx.x;
    if (i < n) p[i] = rsqrtf(p[i]);
}

// x [N,9] fp32 -> xb [N,32] bf16, zero-padded K
__global__ void k_cvt_x(const float* __restrict__ x, __bf16* __restrict__ xb, int N) {
    long idx = (long)blockIdx.x * blockDim.x + threadIdx.x;
    if (idx >= (long)N * 32) return;
    int n = (int)(idx >> 5), c = (int)(idx & 31);
    xb[idx] = (c < 9) ? (__bf16)x[n * 9 + c] : (__bf16)0.0f;
}

// W [Kin, NC] fp32 row-major -> Wt [NC, Kpad] bf16 (column-major, K padded)
__global__ void k_cvt_wt(const float* __restrict__ W, __bf16* __restrict__ Wt,
                         int Kin, int Kpad, int NC) {
    int idx = blockIdx.x * blockDim.x + threadIdx.x;
    if (idx >= NC * Kpad) return;
    int col = idx / Kpad, k = idx % Kpad;
    Wt[idx] = (k < Kin) ? (__bf16)W[k * NC + col] : (__bf16)0.0f;
}

// ---------------- WMMA bf16 GEMM: C[M,NC] = A[M,K] * Bt[NC,K]^T ----------------
// One wave32 per 16x16 output tile, K in steps of 32 via v_wmma_f32_16x16x32_bf16.
__global__ void k_gemm_bf16(const __bf16* __restrict__ A, const __bf16* __restrict__ Bt,
                            float* __restrict__ C, int M, int K, int NC) {
    int wave = threadIdx.x >> 5;
    int lane = threadIdx.x & 31;
    int Nt = NC >> 4;
    int total = (M >> 4) * Nt;
    int tile = blockIdx.x * (blockDim.x >> 5) + wave;
    if (tile >= total) return;
    int tm = tile / Nt, tn = tile % Nt;
    int half = lane >> 4;         // 0: K {0..7,16..23}, 1: K {8..15,24..31}
    int r    = lane & 15;         // row of A / col of B within tile

    const __bf16* arow = A  + (size_t)(tm * 16 + r) * K + half * 8;
    const __bf16* brow = Bt + (size_t)(tn * 16 + r) * K + half * 8;

    v8f acc = {};
    for (int k = 0; k < K; k += 32) {
        v8bf a0 = *(const v8bf*)(arow + k);
        v8bf a1 = *(const v8bf*)(arow + k + 16);
        v8bf b0 = *(const v8bf*)(brow + k);
        v8bf b1 = *(const v8bf*)(brow + k + 16);
        v16bf af, bf;
#pragma unroll
        for (int i = 0; i < 8; ++i) { af[i] = a0[i]; af[i + 8] = a1[i];
                                      bf[i] = b0[i]; bf[i + 8] = b1[i]; }
        acc = __builtin_amdgcn_wmma_f32_16x16x32_bf16(
                  false, af, false, bf, (short)0, acc, false, false);
    }

    // C/D layout: VGPR j -> row (tm*16 + j + half*8), col (tn*16 + r)
    float* cbase = C + (size_t)(tm * 16 + half * 8) * NC + tn * 16 + r;
#pragma unroll
    for (int j = 0; j < 8; ++j) cbase[(size_t)j * NC] = acc[j];
}

// ---------------- edge aggregation (L2-resident gather + atomic scatter) ----------------
// one wave per edge; each lane moves a float4 of the 128-wide feature row
__global__ void k_edge(const int* __restrict__ src, const int* __restrict__ dst,
                       const float* __restrict__ h, const float* __restrict__ dinv,
                       float* __restrict__ a, int E) {
    long gid = (long)blockIdx.x * blockDim.x + threadIdx.x;
    long e = gid >> 5;
    int lane = (int)(gid & 31);
    if (e >= E) return;
    int s = src[e], d = dst[e];
    float norm = dinv[s] * dinv[d];
    float4 v = ((const float4*)(h + (size_t)s * HH))[lane];
    float* ap = a + (size_t)d * HH + lane * 4;
    atomicAdd(ap + 0, v.x * norm);
    atomicAdd(ap + 1, v.y * norm);
    atomicAdd(ap + 2, v.z * norm);
    atomicAdd(ap + 3, v.w * norm);
}

// fused: self-loop + bias + BN(eval) + ReLU; writes fp32 in place and bf16 for next GEMM
__global__ void k_post(float* __restrict__ a, const float* __restrict__ h,
                       const float* __restrict__ dinv,
                       const float* __restrict__ bias, const float* __restrict__ gg,
                       const float* __restrict__ be, const float* __restrict__ rm,
                       const float* __restrict__ rv, __bf16* __restrict__ yb, int N) {
    long idx = (long)blockIdx.x * blockDim.x + threadIdx.x;
    if (idx >= (long)N * HH) return;
    int n = (int)(idx >> 7), c = (int)(idx & 127);
    float di = dinv[n];
    float t = a[idx] + h[idx] * di * di + bias[c];
    t = (t - rm[c]) * rsqrtf(rv[c] + BN_EPS) * gg[c] + be[c];
    t = fmaxf(t, 0.0f);
    a[idx] = t;
    yb[idx] = (__bf16)t;
}

// ---------------- pooling + MLP head ----------------

// one block per graph (contiguous npg-node segments), thread = channel
__global__ void k_pool(const float* __restrict__ y, float* __restrict__ z, int npg) {
    int g = blockIdx.x, c = threadIdx.x;
    const float* p = y + (size_t)g * npg * HH + c;
    float s = 0.0f, m = -3.4e38f;
    for (int i = 0; i < npg; ++i) {
        float v = p[(size_t)i * HH];
        s += v;
        m = fmaxf(m, v);
    }
    z[(size_t)g * 256 + c]       = s / (float)npg;
    z[(size_t)g * 256 + 128 + c] = m;
}

__global__ void k_dense(const float* __restrict__ zin, const float* __restrict__ W,
                        const float* __restrict__ b, float* __restrict__ out,
                        int Gn, int Kin, int Nout, int do_relu) {
    int idx = blockIdx.x * blockDim.x + threadIdx.x;
    if (idx >= Gn * Nout) return;
    int g = idx / Nout, j = idx % Nout;
    float s = b[j];
    const float* zp = zin + (size_t)g * Kin;
    for (int k = 0; k < Kin; ++k) s += zp[k] * W[(size_t)k * Nout + j];
    if (do_relu) s = fmaxf(s, 0.0f);
    out[idx] = s;
}

// ---------------- host launch ----------------

extern "C" void kernel_launch(void* const* d_in, const int* in_sizes, int n_in,
                              void* d_out, int out_size, void* d_ws, size_t ws_size,
                              hipStream_t stream) {
    (void)n_in; (void)ws_size;
    const float* x   = (const float*)d_in[0];
    const int*   ei  = (const int*)d_in[1];
    const float* W1  = (const float*)d_in[3];
    const float* b1  = (const float*)d_in[4];
    const float* W2  = (const float*)d_in[5];
    const float* b2  = (const float*)d_in[6];
    const float* W3  = (const float*)d_in[7];
    const float* b3  = (const float*)d_in[8];
    const float* g1  = (const float*)d_in[9];
    const float* be1 = (const float*)d_in[10];
    const float* rm1 = (const float*)d_in[11];
    const float* rv1 = (const float*)d_in[12];
    const float* g2  = (const float*)d_in[13];
    const float* be2 = (const float*)d_in[14];
    const float* rm2 = (const float*)d_in[15];
    const float* rv2 = (const float*)d_in[16];
    const float* g3  = (const float*)d_in[17];
    const float* be3 = (const float*)d_in[18];
    const float* rm3 = (const float*)d_in[19];
    const float* rv3 = (const float*)d_in[20];
    const float* fw1 = (const float*)d_in[21];
    const float* fb1 = (const float*)d_in[22];
    const float* fw2 = (const float*)d_in[23];
    const float* fb2 = (const float*)d_in[24];
    const float* fw3 = (const float*)d_in[25];
    const float* fb3 = (const float*)d_in[26];

    const int N  = in_sizes[2];        // 100000
    const int E  = in_sizes[1] / 2;    // 1600000
    const int Gn = out_size / 2;       // 1000
    const int npg = N / Gn;            // 100 (contiguous graphs)
    const int* srcp = ei;
    const int* dstp = ei + E;

    // workspace carve-out
    char* ws = (char*)d_ws;
    size_t off = 0;
    auto carve = [&](size_t bytes) {
        size_t o = off;
        off += (bytes + 255) & ~(size_t)255;
        return o;
    };
    float*  dinv = (float*) (ws + carve((size_t)N * 4));
    __bf16* xbf  = (__bf16*)(ws + carve((size_t)N * 32 * 2));
    __bf16* ybf  = (__bf16*)(ws + carve((size_t)N * HH * 2));
    float*  hbuf = (float*) (ws + carve((size_t)N * HH * 4));
    float*  abuf = (float*) (ws + carve((size_t)N * HH * 4));
    __bf16* w1t  = (__bf16*)(ws + carve(128 * 32 * 2));
    __bf16* w2t  = (__bf16*)(ws + carve(128 * 128 * 2));
    __bf16* w3t  = (__bf16*)(ws + carve(128 * 128 * 2));
    float*  zp   = (float*) (ws + carve((size_t)Gn * 256 * 4));
    float*  z1   = (float*) (ws + carve((size_t)Gn * 128 * 4));
    float*  z2   = (float*) (ws + carve((size_t)Gn * 64 * 4));

    const int T = 256;
    const long NH = (long)N * HH;

    // degrees: 1 (self loop) + in-degree, then rsqrt
    k_fill<<<(N + T - 1) / T, T, 0, stream>>>(dinv, 1.0f, N);
    k_deg<<<(E + T - 1) / T, T, 0, stream>>>(dstp, dinv, E);
    k_rsqrt_inplace<<<(N + T - 1) / T, T, 0, stream>>>(dinv, N);

    // weight / input conversions to bf16
    k_cvt_x<<<(int)(((long)N * 32 + T - 1) / T), T, 0, stream>>>(x, xbf, N);
    k_cvt_wt<<<(128 * 32  + T - 1) / T, T, 0, stream>>>(W1, w1t, 9,   32,  128);
    k_cvt_wt<<<(128 * 128 + T - 1) / T, T, 0, stream>>>(W2, w2t, 128, 128, 128);
    k_cvt_wt<<<(128 * 128 + T - 1) / T, T, 0, stream>>>(W3, w3t, 128, 128, 128);

    const int tiles     = (N / 16) * (HH / 16);      // 50000 tiles
    const int gemmGrid  = (tiles + 7) / 8;           // 8 waves per 256-thread block
    const int edgeGrid  = (int)(((long)E * 32 + T - 1) / T);
    const int ewGrid    = (int)((NH + T - 1) / T);

    // ---- layer 1 ----
    k_gemm_bf16<<<gemmGrid, T, 0, stream>>>(xbf, w1t, hbuf, N, 32, HH);
    k_fill<<<ewGrid, T, 0, stream>>>(abuf, 0.0f, NH);
    k_edge<<<edgeGrid, T, 0, stream>>>(srcp, dstp, hbuf, dinv, abuf, E);
    k_post<<<ewGrid, T, 0, stream>>>(abuf, hbuf, dinv, b1, g1, be1, rm1, rv1, ybf, N);

    // ---- layer 2 ----
    k_gemm_bf16<<<gemmGrid, T, 0, stream>>>(ybf, w2t, hbuf, N, 128, HH);
    k_fill<<<ewGrid, T, 0, stream>>>(abuf, 0.0f, NH);
    k_edge<<<edgeGrid, T, 0, stream>>>(srcp, dstp, hbuf, dinv, abuf, E);
    k_post<<<ewGrid, T, 0, stream>>>(abuf, hbuf, dinv, b2, g2, be2, rm2, rv2, ybf, N);

    // ---- layer 3 ----
    k_gemm_bf16<<<gemmGrid, T, 0, stream>>>(ybf, w3t, hbuf, N, 128, HH);
    k_fill<<<ewGrid, T, 0, stream>>>(abuf, 0.0f, NH);
    k_edge<<<edgeGrid, T, 0, stream>>>(srcp, dstp, hbuf, dinv, abuf, E);
    k_post<<<ewGrid, T, 0, stream>>>(abuf, hbuf, dinv, b3, g3, be3, rm3, rv3, ybf, N);

    // ---- pooling + head ----
    k_pool<<<Gn, 128, 0, stream>>>(abuf, zp, npg);
    k_dense<<<(Gn * 128 + T - 1) / T, T, 0, stream>>>(zp, fw1, fb1, z1, Gn, 256, 128, 1);
    k_dense<<<(Gn * 64  + T - 1) / T, T, 0, stream>>>(z1, fw2, fb2, z2, Gn, 128, 64, 1);
    k_dense<<<(Gn * 2   + T - 1) / T, T, 0, stream>>>(z2, fw3, fb3, (float*)d_out, Gn, 64, 2, 0);
}
